// StructureModuleMonomer_35210141892767
// MI455X (gfx1250) — matrix operand order repaired and checked by reading, not compiled, not run on hardware
//
#include <hip/hip_runtime.h>
#include <hip/hip_bf16.h>
#include <math.h>

// ---------------------------------------------------------------------------
// Types for CDNA5 WMMA (wave32)
// ---------------------------------------------------------------------------
typedef __attribute__((ext_vector_type(16))) __bf16 v16bf;
typedef __attribute__((ext_vector_type(8)))  __bf16 v8bf;
typedef __attribute__((ext_vector_type(8)))  float  v8f;
typedef __attribute__((ext_vector_type(4)))  float  v4f;
typedef __attribute__((ext_vector_type(4)))  int    v4i;

#define NRES   512
#define CS     384
#define CZ     128
#define HH     12
#define CIPA   16
#define PQKN   4
#define PVN    8
#define CRES   128
#define CATW   2112           // H*(C_Z + C_IPA + PV*4) = 12*176
#define NBLK   8
#define TSF    10.0f
#define EPSF   1e-8f
#define INFF   100000.0f
#define LNEPS  1e-5f
#define ALPHA_QK 0.14433756729740643f   // sqrt(1/(3*16))

// gfx1250 async global->LDS (guarded: falls back to regular copies if absent)
#if defined(__has_builtin)
#if __has_builtin(__builtin_amdgcn_global_load_async_to_lds_b128)
#define HAVE_ASYNC_LDS 1
#endif
#endif
#ifndef HAVE_ASYNC_LDS
#define HAVE_ASYNC_LDS 0
#endif

// pointer types matching the builtin's signature: v4i in global / LDS AS
typedef __attribute__((address_space(1))) v4i* gp_v4i;
typedef __attribute__((address_space(3))) v4i* lp_v4i;

// float -> bf16, round-to-nearest-even
__device__ __forceinline__ __bf16 f2bf(float f) {
    unsigned u = __builtin_bit_cast(unsigned, f);
    u += 0x7FFFu + ((u >> 16) & 1u);
    unsigned short s = (unsigned short)(u >> 16);
    return __builtin_bit_cast(__bf16, s);
}
__device__ __forceinline__ __bf16 bf_zero() {
    return __builtin_bit_cast(__bf16, (unsigned short)0);
}

// ---------------------------------------------------------------------------
// Flexible batched GEMM: C[b] = act( alpha * A[b] @ B[b] + bias + resid )
// A: [M x K] (lda, batch stride sAb); B: [K x N] (ldb, sBb); C: [M x N].
// ABF/BBF: source already bf16 (else fp32, RNE->bf16 on stage).
// LDS: A row-major (pitch 40 bf16), B stored TRANSPOSED (sBT[n][k], pitch 40)
// so every WMMA fragment is two contiguous 16-byte ds_load_b128.
// bf16 A tiles use GLOBAL_LOAD_ASYNC_TO_LDS_B128 (ASYNCcnt) when available.
// Block = 128 threads = 4 wave32; tile 64(M) x 64(N), K step 32.
// ---------------------------------------------------------------------------
template<int ACTA, int ACTC, int ABF, int BBF>
__global__ __launch_bounds__(128)
void gemm_wmma(const void* __restrict__ Av, const void* __restrict__ Bv,
               const float* __restrict__ bias, const float* __restrict__ resid,
               float* __restrict__ C, int M, int N, int K,
               int lda, int ldb, int ldc,
               long sAb, long sBb, long sCb, float alpha)
{
    __shared__ v8bf sA8[64][5];    // 64 rows x 40 bf16 (cols 0..31 used)
    __shared__ v8bf sB8[64][5];    // B^T: 64 n-rows x 40 bf16 (k 0..31 used)
    __bf16* sAe = (__bf16*)sA8;
    __bf16* sBe = (__bf16*)sB8;

    const int tid  = threadIdx.x;
    const int wave = tid >> 5;
    const int lane = tid & 31;
    const int half = lane >> 4;
    const int lr   = lane & 15;
    const int tileM = blockIdx.y * 64;
    const int tileN = blockIdx.x * 64;
    const long bo = blockIdx.z;

    const float*  Af = (const float*)Av  + (ABF ? 0 : bo * sAb);
    const __bf16* Ab = (const __bf16*)Av + (ABF ? bo * sAb : 0);
    const float*  Bf = (const float*)Bv  + (BBF ? 0 : bo * sBb);
    const __bf16* Bb = (const __bf16*)Bv + (BBF ? bo * sBb : 0);
    C += bo * sCb;
    const float* Rr = resid ? resid + bo * sCb : nullptr;

    const bool kWhole = (K % 32) == 0;
    const bool fastA = kWhole && (tileM + 64 <= M) && (lda % (ABF ? 8 : 4) == 0);
    const bool fastB = kWhole && (tileN + 64 <= N) && (ldb % (BBF ? 8 : 4) == 0);

    v8f acc[4] = {};

    for (int k0 = 0; k0 < K; k0 += 32) {
        // ---------------- stage A tile 64x32 ----------------
        if (fastA) {
            const int r = tid >> 1, cb = (tid & 1) * 16;
            if (ABF) {
                const __bf16* src = Ab + (long)(tileM + r) * lda + k0 + cb;
#if HAVE_ASYNC_LDS
                // async LDS-DMA: 2 x 16B per thread, tracked on ASYNCcnt
                __builtin_amdgcn_global_load_async_to_lds_b128(
                    (gp_v4i)(const void*)src,
                    (lp_v4i)(void*)&sA8[r][(cb >> 3) + 0], 0, 0);
                __builtin_amdgcn_global_load_async_to_lds_b128(
                    (gp_v4i)(const void*)(src + 8),
                    (lp_v4i)(void*)&sA8[r][(cb >> 3) + 1], 0, 0);
#else
                sA8[r][(cb >> 3) + 0] = *(const v8bf*)(src + 0);
                sA8[r][(cb >> 3) + 1] = *(const v8bf*)(src + 8);
#endif
            } else {
                const float* src = Af + (long)(tileM + r) * lda + k0 + cb;
                #pragma unroll
                for (int j = 0; j < 2; ++j) {
                    v4f x0 = *(const v4f*)(src + j * 8);
                    v4f x1 = *(const v4f*)(src + j * 8 + 4);
                    v8bf o;
                    #pragma unroll
                    for (int e = 0; e < 4; ++e) {
                        float a = x0[e], b = x1[e];
                        if (ACTA) { a = fmaxf(a, 0.f); b = fmaxf(b, 0.f); }
                        o[e] = f2bf(a); o[e + 4] = f2bf(b);
                    }
                    sA8[r][(cb >> 3) + j] = o;
                }
            }
        } else {
            for (int i = tid; i < 64 * 32; i += 128) {
                int r = i >> 5, c = i & 31;
                int gm = tileM + r, gk = k0 + c;
                __bf16 bv = bf_zero();
                if (gm < M && gk < K) {
                    long idx = (long)gm * lda + gk;
                    if (ABF) bv = Ab[idx];
                    else {
                        float v = Af[idx];
                        if (ACTA) v = fmaxf(v, 0.f);
                        bv = f2bf(v);
                    }
                }
                sAe[r * 40 + c] = bv;
            }
        }
        // ---------------- stage B tile (transposed into LDS) ----------------
        if (fastB) {
            const int r = tid >> 2, nb = (tid & 3) * 16;
            if (BBF) {
                const __bf16* src = Bb + (long)(k0 + r) * ldb + tileN + nb;
                v8bf x0 = *(const v8bf*)(src);
                v8bf x1 = *(const v8bf*)(src + 8);
                #pragma unroll
                for (int e = 0; e < 8; ++e) {
                    sBe[(nb + e) * 40 + r]     = x0[e];
                    sBe[(nb + 8 + e) * 40 + r] = x1[e];
                }
            } else {
                const float* src = Bf + (long)(k0 + r) * ldb + tileN + nb;
                #pragma unroll
                for (int j = 0; j < 4; ++j) {
                    v4f xv = *(const v4f*)(src + j * 4);
                    #pragma unroll
                    for (int e = 0; e < 4; ++e)
                        sBe[(nb + j * 4 + e) * 40 + r] = f2bf(xv[e]);
                }
            }
        } else {
            for (int i = tid; i < 64 * 32; i += 128) {
                int r = i >> 6, c = i & 63;     // r = k index, c = n index
                int gk = k0 + r, gn = tileN + c;
                __bf16 bv = bf_zero();
                if (gk < K && gn < N) {
                    long idx = (long)gk * ldb + gn;
                    if (BBF) bv = Bb[idx];
                    else     bv = f2bf(Bf[idx]);
                }
                sBe[c * 40 + r] = bv;
            }
        }
        // prefetch next A K-tile (gfx1250 global_prefetch path)
        if (k0 + 32 < K) {
            long roff = (long)(tileM + (tid >> 1)) * lda + k0 + 32;
            __builtin_prefetch(ABF ? (const void*)(Ab + roff)
                                   : (const void*)(Af + roff), 0, 1);
        }
#if HAVE_ASYNC_LDS
        if (ABF && fastA) {
#if defined(__has_builtin) && __has_builtin(__builtin_amdgcn_s_wait_asynccnt)
            __builtin_amdgcn_s_wait_asynccnt(0);
#else
            asm volatile("s_wait_asynccnt 0" ::: "memory");
#endif
        }
#endif
        __syncthreads();

        // A fragment (ISA 7.12.2): K = half*8 + 0..7 | 16+half*8 + 0..7
        const int arow = wave * 16 + lr;
        v8bf alo = sA8[arow][half];
        v8bf ahi = sA8[arow][2 + half];
        v16bf af = __builtin_shufflevector(alo, ahi,
            0,1,2,3,4,5,6,7,8,9,10,11,12,13,14,15);

        #pragma unroll
        for (int nb = 0; nb < 4; ++nb) {
            const int n = nb * 16 + lr;
            v8bf b0 = sB8[n][half];
            v8bf b1 = sB8[n][2 + half];
            v16bf bfv = __builtin_shufflevector(b0, b1,
                0,1,2,3,4,5,6,7,8,9,10,11,12,13,14,15);
            acc[nb] = __builtin_amdgcn_wmma_f32_16x16x32_bf16(
                false, af, false, bfv, (short)0, acc[nb], false, false);
        }
        __syncthreads();
    }

    // epilogue: C/D layout — VGPR r holds M = r (lanes 0-15) / r+8 (lanes 16-31)
    #pragma unroll
    for (int nb = 0; nb < 4; ++nb) {
        int gn = tileN + nb * 16 + lr;
        float bv = (bias && gn < N) ? bias[gn] : 0.f;
        #pragma unroll
        for (int r = 0; r < 8; ++r) {
            int gm = tileM + wave * 16 + (half ? 8 : 0) + r;
            if (gm < M && gn < N) {
                float v = acc[nb][r] * alpha + bv;
                if (Rr)   v += Rr[(long)gm * ldc + gn];
                if (ACTC) v = fmaxf(v, 0.f);
                C[(long)gm * ldc + gn] = v;
            }
        }
    }
}

// ---------------------------------------------------------------------------
// LayerNorm: one 128-thread block per row; optional bf16 output
// ---------------------------------------------------------------------------
template<int OBF>
__global__ __launch_bounds__(128)
void k_layernorm(const float* __restrict__ x, const float* __restrict__ g,
                 const float* __restrict__ b, void* __restrict__ y, int cols)
{
    const long row = blockIdx.x;
    const float* xr = x + row * cols;
    const int tid = threadIdx.x;
    __shared__ float red[128];

    float v[3]; int cnt = 0; float s = 0.f;
    for (int c = tid; c < cols; c += 128) { float t = xr[c]; v[cnt++] = t; s += t; }
    red[tid] = s; __syncthreads();
    for (int st = 64; st > 0; st >>= 1) { if (tid < st) red[tid] += red[tid + st]; __syncthreads(); }
    const float mean = red[0] / cols; __syncthreads();

    float vs = 0.f;
    for (int k = 0; k < cnt; ++k) { float d = v[k] - mean; vs += d * d; }
    red[tid] = vs; __syncthreads();
    for (int st = 64; st > 0; st >>= 1) { if (tid < st) red[tid] += red[tid + st]; __syncthreads(); }
    const float inv = rsqrtf(red[0] / cols + LNEPS); __syncthreads();

    cnt = 0;
    for (int c = tid; c < cols; c += 128) {
        float o = (v[cnt++] - mean) * inv * g[c] + b[c];
        if (OBF) ((__bf16*)y)[row * cols + c] = f2bf(o);
        else     ((float*)y)[row * cols + c] = o;
    }
}

// ---------------------------------------------------------------------------
// quaternion helpers
// ---------------------------------------------------------------------------
__device__ __forceinline__ void quat2rot(const float q[4], float R[9]) {
    float w = q[0], x = q[1], y = q[2], z = q[3];
    R[0] = 1.f - 2.f * (y * y + z * z); R[1] = 2.f * (x * y - w * z); R[2] = 2.f * (x * z + w * y);
    R[3] = 2.f * (x * y + w * z); R[4] = 1.f - 2.f * (x * x + z * z); R[5] = 2.f * (y * z - w * x);
    R[6] = 2.f * (x * z - w * y); R[7] = 2.f * (y * z + w * x); R[8] = 1.f - 2.f * (x * x + y * y);
}
__device__ __forceinline__ void quat_mul(const float a[4], const float b[4], float o[4]) {
    o[0] = a[0]*b[0] - a[1]*b[1] - a[2]*b[2] - a[3]*b[3];
    o[1] = a[0]*b[1] + a[1]*b[0] + a[2]*b[3] - a[3]*b[2];
    o[2] = a[0]*b[2] - a[1]*b[3] + a[2]*b[0] + a[3]*b[1];
    o[3] = a[0]*b[3] + a[1]*b[2] - a[2]*b[1] + a[3]*b[0];
}

__global__ void k_init_frames(float* quat, float* trans) {
    int i = blockIdx.x * blockDim.x + threadIdx.x;
    if (i < NRES) {
        quat[i*4+0] = 1.f; quat[i*4+1] = 0.f; quat[i*4+2] = 0.f; quat[i*4+3] = 0.f;
        trans[i*3+0] = 0.f; trans[i*3+1] = 0.f; trans[i*3+2] = 0.f;
    }
}

// Rotate q/k/v points into global frame
__global__ __launch_bounds__(64)
void k_points(const float* __restrict__ qp_raw, const float* __restrict__ kvp_raw,
              const float* __restrict__ quat, const float* __restrict__ trans,
              float* __restrict__ qp_rot, float* __restrict__ kp_rot,
              float* __restrict__ vp_rot)
{
    const int i = blockIdx.x;
    const int tid = threadIdx.x;
    float q[4] = {quat[i*4], quat[i*4+1], quat[i*4+2], quat[i*4+3]};
    float R[9]; quat2rot(q, R);
    float t0 = trans[i*3], t1 = trans[i*3+1], t2 = trans[i*3+2];

    for (int idx = tid; idx < HH * PQKN; idx += 64) {      // qp: 48 points
        float px = qp_raw[i*144 + idx], py = qp_raw[i*144 + 48 + idx], pz = qp_raw[i*144 + 96 + idx];
        qp_rot[i*144 + idx*3 + 0] = R[0]*px + R[1]*py + R[2]*pz + t0;
        qp_rot[i*144 + idx*3 + 1] = R[3]*px + R[4]*py + R[5]*pz + t1;
        qp_rot[i*144 + idx*3 + 2] = R[6]*px + R[7]*py + R[8]*pz + t2;
    }
    for (int idx = tid; idx < HH * (PQKN + PVN); idx += 64) { // kvp: 144 points
        float px = kvp_raw[i*432 + idx], py = kvp_raw[i*432 + 144 + idx], pz = kvp_raw[i*432 + 288 + idx];
        float rx = R[0]*px + R[1]*py + R[2]*pz + t0;
        float ry = R[3]*px + R[4]*py + R[5]*pz + t1;
        float rz = R[6]*px + R[7]*py + R[8]*pz + t2;
        int h = idx / 12, pp = idx % 12;
        if (pp < PQKN) {
            int o = i*144 + (h*PQKN + pp)*3;
            kp_rot[o] = rx; kp_rot[o+1] = ry; kp_rot[o+2] = rz;
        } else {
            int o = i*288 + (h*PVN + pp - PQKN)*3;
            vp_rot[o] = rx; vp_rot[o+1] = ry; vp_rot[o+2] = rz;
        }
    }
}

// Fused attention row kernel: logits = alpha*q.k + sqrt(1/3)*b + pt + mask,
// softmax over j, emit bf16 attention row. One block per (i,h).
__global__ __launch_bounds__(256)
void k_attn_softmax(const float* __restrict__ qb, const float* __restrict__ kv,
                    __bf16* __restrict__ attn16, const float* __restrict__ bmat,
                    const float* __restrict__ qp_rot, const float* __restrict__ kp_rot,
                    const float* __restrict__ head_w, const float* __restrict__ mask)
{
    const int i = blockIdx.x, h = blockIdx.y, tid = threadIdx.x;
    __shared__ float red[256];
    __shared__ float qs[16];
    __shared__ float qp[12];
    __shared__ float sh_hw, sh_mi;
    if (tid < 16) qs[tid] = qb[i * (HH * CIPA) + h * CIPA + tid];
    if (tid < 12) qp[tid] = qp_rot[i*144 + h*12 + tid];
    if (tid == 0) {
        sh_hw = log1pf(expf(head_w[h])) * 0.13608276348795434f;  // sqrt(1/54)
        sh_mi = mask[i];
    }
    __syncthreads();

    __bf16* arow = attn16 + ((long)h * NRES + i) * NRES;
    float lv[2]; float mx = -3.0e38f;
    #pragma unroll
    for (int r = 0; r < 2; ++r) {
        int j = tid + r * 256;
        // q . k  (k[j,h,c] = kv[j*384 + h*32 + c])
        const float* krow = kv + (long)j * (2 * HH * CIPA) + h * 2 * CIPA;
        float qk = 0.f;
        #pragma unroll
        for (int c = 0; c < CIPA; ++c) qk += qs[c] * krow[c];
        // point-distance term
        const float* kp = kp_rot + j*144 + h*12;
        float d2 = 0.f;
        #pragma unroll
        for (int p = 0; p < 4; ++p) {
            float dx = qp[p*3+0] - kp[p*3+0];
            float dy = qp[p*3+1] - kp[p*3+1];
            float dz = qp[p*3+2] - kp[p*3+2];
            d2 += dx*dx + dy*dy + dz*dz;
        }
        float lg = ALPHA_QK * qk
                 + 0.5773502691896258f * bmat[((long)i * NRES + j) * HH + h]
                 - 0.5f * sh_hw * d2
                 + INFF * (sh_mi * mask[j] - 1.f);
        lv[r] = lg; mx = fmaxf(mx, lg);
    }
    red[tid] = mx; __syncthreads();
    for (int s = 128; s > 0; s >>= 1) { if (tid < s) red[tid] = fmaxf(red[tid], red[tid+s]); __syncthreads(); }
    mx = red[0]; __syncthreads();
    float sum = 0.f;
    #pragma unroll
    for (int r = 0; r < 2; ++r) { lv[r] = expf(lv[r] - mx); sum += lv[r]; }
    red[tid] = sum; __syncthreads();
    for (int s = 128; s > 0; s >>= 1) { if (tid < s) red[tid] += red[tid+s]; __syncthreads(); }
    const float inv = 1.f / red[0];
    #pragma unroll
    for (int r = 0; r < 2; ++r) arow[tid + r * 256] = f2bf(lv[r] * inv);
}

// op -> local frame (R^T(op - t)), write into cat[192..576)
__global__ __launch_bounds__(128)
void k_cat_op(const float* __restrict__ op_buf, const float* __restrict__ quat,
              const float* __restrict__ trans, float* __restrict__ cat)
{
    const int i = blockIdx.x, tid = threadIdx.x;
    float q[4] = {quat[i*4], quat[i*4+1], quat[i*4+2], quat[i*4+3]};
    float R[9]; quat2rot(q, R);
    float t0 = trans[i*3], t1 = trans[i*3+1], t2 = trans[i*3+2];
    if (tid < HH * PVN) {
        int h = tid >> 3, p = tid & 7;
        const float* o = op_buf + i*288 + h*24 + p*3;
        float vx = o[0] - t0, vy = o[1] - t1, vz = o[2] - t2;
        float lx = R[0]*vx + R[3]*vy + R[6]*vz;
        float ly = R[1]*vx + R[4]*vy + R[7]*vz;
        float lz = R[2]*vx + R[5]*vy + R[8]*vz;
        float* c = cat + (long)i * CATW;
        c[192 + tid] = lx;
        c[288 + tid] = ly;
        c[384 + tid] = lz;
        c[480 + tid] = sqrtf(lx*lx + ly*ly + lz*lz + EPSF);
    }
}

// backbone update: trans += R @ upd[3:], quat = normalize(quat * (1,upd[:3]))
__global__ __launch_bounds__(64)
void k_bb_update(float* __restrict__ quat, float* __restrict__ trans,
                 const float* __restrict__ upd)
{
    int i = blockIdx.x * 64 + threadIdx.x;
    if (i >= NRES) return;
    float q[4] = {quat[i*4], quat[i*4+1], quat[i*4+2], quat[i*4+3]};
    float R[9]; quat2rot(q, R);
    const float* u = upd + i*6;
    trans[i*3+0] += R[0]*u[3] + R[1]*u[4] + R[2]*u[5];
    trans[i*3+1] += R[3]*u[3] + R[4]*u[4] + R[5]*u[5];
    trans[i*3+2] += R[6]*u[3] + R[7]*u[4] + R[8]*u[5];
    float dq[4] = {1.f, u[0], u[1], u[2]};
    float nq[4]; quat_mul(q, dq, nq);
    float n = rsqrtf(nq[0]*nq[0] + nq[1]*nq[1] + nq[2]*nq[2] + nq[3]*nq[3]);
    quat[i*4+0] = nq[0]*n; quat[i*4+1] = nq[1]*n; quat[i*4+2] = nq[2]*n; quat[i*4+3] = nq[3]*n;
}

// torsion->frames, frames->positions, all per-iteration outputs
__global__ __launch_bounds__(64)
void k_finalize(const float* __restrict__ quat, const float* __restrict__ trans,
                const float* __restrict__ un_raw, const int* __restrict__ aatype,
                const float* __restrict__ def_frames, const int* __restrict__ group_idx,
                const float* __restrict__ atom_mask, const float* __restrict__ lit_pos,
                float* __restrict__ out, int it,
                long off_frames, long off_sc, long off_unn, long off_angs, long off_poss)
{
    int i = blockIdx.x * 64 + threadIdx.x;
    if (i >= NRES) return;
    float q[4] = {quat[i*4], quat[i*4+1], quat[i*4+2], quat[i*4+3]};
    float tb[3] = {trans[i*3]*TSF, trans[i*3+1]*TSF, trans[i*3+2]*TSF};
    float Rb[9]; quat2rot(q, Rb);

    float* fo = out + off_frames + ((long)it * NRES + i) * 7;
    fo[0]=q[0]; fo[1]=q[1]; fo[2]=q[2]; fo[3]=q[3]; fo[4]=tb[0]; fo[5]=tb[1]; fo[6]=tb[2];

    float al[8][2]; al[0][0] = 0.f; al[0][1] = 1.f;
    float* uo = out + off_unn  + ((long)it * NRES + i) * 14;
    float* ao = out + off_angs + ((long)it * NRES + i) * 14;
    for (int a = 0; a < 7; ++a) {
        float ux = un_raw[i*14 + a*2], uy = un_raw[i*14 + a*2 + 1];
        float n = sqrtf(fmaxf(ux*ux + uy*uy, EPSF));
        uo[a*2] = ux; uo[a*2+1] = uy;
        ao[a*2] = ux/n; ao[a*2+1] = uy/n;
        al[a+1][0] = ux/n; al[a+1][1] = uy/n;
    }

    const int ai = aatype[i];
    float Rl[8][9], tl[8][3];
    for (int g = 0; g < 8; ++g) {
        const float* d = def_frames + (long)(ai*8 + g) * 16;
        float sa = al[g][0], ca = al[g][1];
        for (int r = 0; r < 3; ++r) {
            float d0 = d[r*4+0], d1 = d[r*4+1], d2 = d[r*4+2];
            Rl[g][r*3+0] = d0;
            Rl[g][r*3+1] = ca*d1 + sa*d2;
            Rl[g][r*3+2] = -sa*d1 + ca*d2;
        }
        tl[g][0] = d[3]; tl[g][1] = d[7]; tl[g][2] = d[11];
    }
    for (int g = 5; g < 8; ++g) {
        int p = g - 1;
        float Rt[9], tt[3];
        for (int r = 0; r < 3; ++r) {
            for (int c = 0; c < 3; ++c)
                Rt[r*3+c] = Rl[p][r*3+0]*Rl[g][0*3+c] + Rl[p][r*3+1]*Rl[g][1*3+c] + Rl[p][r*3+2]*Rl[g][2*3+c];
            tt[r] = Rl[p][r*3+0]*tl[g][0] + Rl[p][r*3+1]*tl[g][1] + Rl[p][r*3+2]*tl[g][2] + tl[p][r];
        }
        for (int k = 0; k < 9; ++k) Rl[g][k] = Rt[k];
        for (int k = 0; k < 3; ++k) tl[g][k] = tt[k];
    }
    float Rg[8][9], tg[8][3];
    float* sco = out + off_sc + ((long)it * NRES + i) * 128;
    for (int g = 0; g < 8; ++g) {
        for (int r = 0; r < 3; ++r) {
            for (int c = 0; c < 3; ++c)
                Rg[g][r*3+c] = Rb[r*3+0]*Rl[g][0*3+c] + Rb[r*3+1]*Rl[g][1*3+c] + Rb[r*3+2]*Rl[g][2*3+c];
            tg[g][r] = Rb[r*3+0]*tl[g][0] + Rb[r*3+1]*tl[g][1] + Rb[r*3+2]*tl[g][2] + tb[r];
        }
        float* f44 = sco + g*16;
        for (int r = 0; r < 3; ++r) {
            f44[r*4+0] = Rg[g][r*3+0]; f44[r*4+1] = Rg[g][r*3+1];
            f44[r*4+2] = Rg[g][r*3+2]; f44[r*4+3] = tg[g][r];
        }
        f44[12] = 0.f; f44[13] = 0.f; f44[14] = 0.f; f44[15] = 1.f;
    }
    float* po = out + off_poss + ((long)it * NRES + i) * 42;
    for (int a = 0; a < 14; ++a) {
        int g = group_idx[ai*14 + a];
        const float* lp = lit_pos + (long)(ai*14 + a) * 3;
        float m = atom_mask[ai*14 + a];
        for (int r = 0; r < 3; ++r)
            po[a*3+r] = (Rg[g][r*3+0]*lp[0] + Rg[g][r*3+1]*lp[1] + Rg[g][r*3+2]*lp[2] + tg[g][r]) * m;
    }
}

// ---------------------------------------------------------------------------
// Host-side orchestration
// ---------------------------------------------------------------------------
static void gemm(hipStream_t st, int AA, int AC, int ABF, int BBF,
                 const void* A, const void* B, const float* bias, const float* resid,
                 float* C, int M, int N, int K, int lda, int ldb, int ldc,
                 long sA, long sB, long sC, int batch, float alpha)
{
    dim3 g((N + 63) / 64, (M + 63) / 64, batch), blk(128);
#define GCASE(aa, ac, ab, bb) \
    if (AA == aa && AC == ac && ABF == ab && BBF == bb) { \
        gemm_wmma<aa, ac, ab, bb><<<g, blk, 0, st>>>(A, B, bias, resid, C, M, N, K, lda, ldb, ldc, sA, sB, sC, alpha); \
        return; }
    GCASE(0,0,0,0) GCASE(0,1,0,0) GCASE(1,0,0,0) GCASE(1,1,0,0)
    GCASE(0,0,1,0) GCASE(0,0,1,1)
#undef GCASE
}

enum {
    IN_SINGLE = 0, IN_PAIR, IN_AATYPE, IN_MASK,
    P_LN_S_G, P_LN_S_B, P_LN_Z_G, P_LN_Z_B,
    P_IN_W, P_IN_B, P_LN_IPA_G, P_LN_IPA_B, P_BB_W, P_BB_B,
    P_Q_W, P_Q_B, P_KV_W, P_KV_B, P_QP_W, P_QP_B, P_KVP_W, P_KVP_B,
    P_B_W, P_B_B, P_HEAD_W, P_OUT_W, P_OUT_B,
    P_T_W1, P_T_B1, P_T_W2, P_T_B2, P_T_W3, P_T_B3, P_T_LN_G, P_T_LN_B,
    P_A_IN_W, P_A_IN_B, P_A_INIT_W, P_A_INIT_B,
    P_A0_W1, P_A0_B1, P_A0_W2, P_A0_B2,
    P_A1_W1, P_A1_B1, P_A1_W2, P_A1_B2,
    P_A_OUT_W, P_A_OUT_B,
    IN_GROUP_IDX, C_DEF_FRAMES, C_ATOM_MASK, C_LIT_POS
};

extern "C" void kernel_launch(void* const* d_in, const int* in_sizes, int n_in,
                              void* d_out, int out_size, void* d_ws, size_t ws_size,
                              hipStream_t stream)
{
    (void)in_sizes; (void)out_size; (void)ws_size;
    const float* F[64];
    for (int k = 0; k < n_in && k < 64; ++k) F[k] = (const float*)d_in[k];
    const int* aatype   = (const int*)d_in[IN_AATYPE];
    const int* groupIdx = (const int*)d_in[IN_GROUP_IDX];
    float* out = (float*)d_out;

    const long OF_FRAMES = 0;
    const long OF_SC     = OF_FRAMES + (long)NBLK*NRES*7;
    const long OF_UNN    = OF_SC     + (long)NBLK*NRES*128;
    const long OF_ANGS   = OF_UNN    + (long)NBLK*NRES*14;
    const long OF_POSS   = OF_ANGS   + (long)NBLK*NRES*14;
    const long OF_STATES = OF_POSS   + (long)NBLK*NRES*42;
    const long OF_S      = OF_STATES + (long)NBLK*NRES*CS;

    float* w = (float*)d_ws;
    size_t off = 0;
    auto alloc = [&](size_t n) { float* p = w + off; off += n; return p; };
    __bf16* z16    = (__bf16*)alloc((size_t)NRES*NRES*CZ/2);   // 67MB bf16
    float* s0      = alloc((size_t)NRES*CS);
    float* s       = alloc((size_t)NRES*CS);
    float* qb      = alloc((size_t)NRES*HH*CIPA);
    float* kv      = alloc((size_t)NRES*2*HH*CIPA);
    float* qp_raw  = alloc((size_t)NRES*144);
    float* kvp_raw = alloc((size_t)NRES*432);
    float* qp_rot  = alloc((size_t)NRES*144);
    float* kp_rot  = alloc((size_t)NRES*144);
    float* vp_rot  = alloc((size_t)NRES*288);
    float* bmat    = alloc((size_t)NRES*NRES*HH);      // fp32
    __bf16* attn16 = (__bf16*)alloc((size_t)HH*NRES*NRES/2);
    float* op_buf  = alloc((size_t)NRES*288);
    float* cat     = alloc((size_t)NRES*CATW);
    float* quat    = alloc((size_t)NRES*4);
    float* trans   = alloc((size_t)NRES*3);
    float* upd     = alloc((size_t)NRES*6);
    float* t1      = alloc((size_t)NRES*CS);
    float* t2      = alloc((size_t)NRES*CS);
    float* initterm= alloc((size_t)NRES*CRES);
    float* abuf    = alloc((size_t)NRES*CRES);
    float* hang    = alloc((size_t)NRES*CRES);
    float* un      = alloc((size_t)NRES*14);

    // ---- prologue ----
    k_layernorm<0><<<NRES, 128, 0, stream>>>(F[IN_SINGLE], F[P_LN_S_G], F[P_LN_S_B], s0, CS);
    k_layernorm<1><<<NRES*NRES, 128, 0, stream>>>(F[IN_PAIR], F[P_LN_Z_G], F[P_LN_Z_B], z16, CZ);
    gemm(stream, 0,0,0,0, s0, F[P_IN_W], F[P_IN_B], nullptr, s, NRES, CS, CS, CS, CS, CS, 0,0,0, 1, 1.f);
    gemm(stream, 1,0,0,0, s0, F[P_A_INIT_W], F[P_A_INIT_B], nullptr, initterm, NRES, CRES, CS, CS, CRES, CRES, 0,0,0, 1, 1.f);
    k_init_frames<<<(NRES+63)/64, 64, 0, stream>>>(quat, trans);
    // pair bias projection b = z @ b_w  (z fixed across iterations)
    gemm(stream, 0,0,1,0, z16, F[P_B_W], F[P_B_B], nullptr, bmat, NRES*NRES, HH, CZ, CZ, HH, HH, 0,0,0, 1, 1.f);

    for (int it = 0; it < NBLK; ++it) {
        // ---- IPA projections ----
        gemm(stream, 0,0,0,0, s, F[P_Q_W],  F[P_Q_B],  nullptr, qb,      NRES, HH*CIPA,   CS, CS, HH*CIPA,   HH*CIPA,   0,0,0, 1, 1.f);
        gemm(stream, 0,0,0,0, s, F[P_KV_W], F[P_KV_B], nullptr, kv,      NRES, 2*HH*CIPA, CS, CS, 2*HH*CIPA, 2*HH*CIPA, 0,0,0, 1, 1.f);
        gemm(stream, 0,0,0,0, s, F[P_QP_W], F[P_QP_B], nullptr, qp_raw,  NRES, 144, CS, CS, 144, 144, 0,0,0, 1, 1.f);
        gemm(stream, 0,0,0,0, s, F[P_KVP_W],F[P_KVP_B],nullptr, kvp_raw, NRES, 432, CS, CS, 432, 432, 0,0,0, 1, 1.f);
        k_points<<<NRES, 64, 0, stream>>>(qp_raw, kvp_raw, quat, trans, qp_rot, kp_rot, vp_rot);

        // fused attention: q.k dot + pair bias + point term + mask + softmax
        k_attn_softmax<<<dim3(NRES, HH), 256, 0, stream>>>(qb, kv, attn16, bmat,
                                                           qp_rot, kp_rot, F[P_HEAD_W], F[IN_MASK]);

        // o = a @ v  -> cat[:, h*16 + c]
        gemm(stream, 0,0,1,0, attn16, kv + CIPA, nullptr, nullptr, cat,
             NRES, CIPA, NRES, NRES, 2*HH*CIPA, CATW,
             (long)NRES*NRES, 2*CIPA, CIPA, HH, 1.f);
        // op = a @ vp_rot -> op_buf[i][h][24]
        gemm(stream, 0,0,1,0, attn16, vp_rot, nullptr, nullptr, op_buf,
             NRES, PVN*3, NRES, NRES, 288, 288,
             (long)NRES*NRES, 24, 24, HH, 1.f);
        // opair: per residue i, A[h][j]=attn[h,i,j] (12x512) @ z[i] (512x128)
        gemm(stream, 0,0,1,1, attn16, z16, nullptr, nullptr, cat + 576,
             HH, CZ, NRES, NRES*NRES, CZ, CZ,
             (long)NRES, (long)NRES*CZ, (long)CATW, NRES, 1.f);
        k_cat_op<<<NRES, 128, 0, stream>>>(op_buf, quat, trans, cat);

        // out projection with residual into s
        gemm(stream, 0,0,0,0, cat, F[P_OUT_W], F[P_OUT_B], s, s, NRES, CS, CATW, CATW, CS, CS, 0,0,0, 1, 1.f);
        k_layernorm<0><<<NRES, 128, 0, stream>>>(s, F[P_LN_IPA_G], F[P_LN_IPA_B], s, CS);

        // ---- transition ----
        gemm(stream, 0,1,0,0, s,  F[P_T_W1], F[P_T_B1], nullptr, t1, NRES, CS, CS, CS, CS, CS, 0,0,0, 1, 1.f);
        gemm(stream, 0,1,0,0, t1, F[P_T_W2], F[P_T_B2], nullptr, t2, NRES, CS, CS, CS, CS, CS, 0,0,0, 1, 1.f);
        gemm(stream, 0,0,0,0, t2, F[P_T_W3], F[P_T_B3], s,       s,  NRES, CS, CS, CS, CS, CS, 0,0,0, 1, 1.f);
        k_layernorm<0><<<NRES, 128, 0, stream>>>(s, F[P_T_LN_G], F[P_T_LN_B], s, CS);

        // ---- backbone update ----
        gemm(stream, 0,0,0,0, s, F[P_BB_W], F[P_BB_B], nullptr, upd, NRES, 6, CS, CS, 6, 6, 0,0,0, 1, 1.f);
        k_bb_update<<<(NRES+63)/64, 64, 0, stream>>>(quat, trans, upd);

        // ---- angle resnet ----
        gemm(stream, 1,0,0,0, s, F[P_A_IN_W], F[P_A_IN_B], initterm, abuf, NRES, CRES, CS, CS, CRES, CRES, 0,0,0, 1, 1.f);
        gemm(stream, 1,1,0,0, abuf, F[P_A0_W1], F[P_A0_B1], nullptr, hang, NRES, CRES, CRES, CRES, CRES, CRES, 0,0,0, 1, 1.f);
        gemm(stream, 0,0,0,0, hang, F[P_A0_W2], F[P_A0_B2], abuf,    abuf, NRES, CRES, CRES, CRES, CRES, CRES, 0,0,0, 1, 1.f);
        gemm(stream, 1,1,0,0, abuf, F[P_A1_W1], F[P_A1_B1], nullptr, hang, NRES, CRES, CRES, CRES, CRES, CRES, 0,0,0, 1, 1.f);
        gemm(stream, 0,0,0,0, hang, F[P_A1_W2], F[P_A1_B2], abuf,    abuf, NRES, CRES, CRES, CRES, CRES, CRES, 0,0,0, 1, 1.f);
        gemm(stream, 1,0,0,0, abuf, F[P_A_OUT_W], F[P_A_OUT_B], nullptr, un, NRES, 14, CRES, CRES, 14, 14, 0,0,0, 1, 1.f);

        // ---- frames / positions / per-iteration outputs ----
        k_finalize<<<(NRES+63)/64, 64, 0, stream>>>(quat, trans, un, aatype,
            F[C_DEF_FRAMES], groupIdx, F[C_ATOM_MASK], F[C_LIT_POS],
            out, it, OF_FRAMES, OF_SC, OF_UNN, OF_ANGS, OF_POSS);
        (void)hipMemcpyAsync(out + OF_STATES + (long)it*NRES*CS, s,
                       (size_t)NRES*CS*sizeof(float), hipMemcpyDeviceToDevice, stream);
    }
    (void)hipMemcpyAsync(out + OF_S, s, (size_t)NRES*CS*sizeof(float),
                   hipMemcpyDeviceToDevice, stream);
}
